// RNN_61735859913526
// MI455X (gfx1250) — matrix operand order repaired and checked by reading, not compiled, not run on hardware
//
#include <hip/hip_runtime.h>
#include <math.h>

// MI455X (gfx1250) persistent-RNN implementation.
// - All GEMMs via v_wmma_f32_16x16x32_f16 (fp32 accumulate).
// - Recurrence: 32 persistent blocks, W_hh slice resident in LDS (64KB/block,
//   staged via async-to-LDS when available), device-scope atomic grid barrier
//   between the 512 sequential steps, double-buffered f16 hidden state.
// - xw precomputed in t-major layout so each step streams one contiguous
//   128KB L2-resident slab (prefetched one step ahead).

typedef __attribute__((ext_vector_type(16))) _Float16 v16h;
typedef __attribute__((ext_vector_type(8)))  _Float16 v8h;
typedef __attribute__((ext_vector_type(8)))  float    v8f;
typedef int v4i __attribute__((vector_size(16)));   // matches async-LDS builtin param

#define B_   64
#define T_   512
#define IN_  256
#define H_   1024
#define OUT_ 128
#define M_   (B_ * T_)   // 32768

#define GLOBAL_AS __attribute__((address_space(1)))
#define LDS_AS    __attribute__((address_space(3)))

// ---------------------------------------------------------------------------
// A-fragment loader for 16-bit WMMA (16x32 tile, MxK).
// Per ISA layout: lanes 0-15 hold M=0..15; VGPR0..3 = K {0..7} (lanes 0-15)
// or K {8..15} (lanes 16-31); VGPR4..7 = K {16..23} / {24..31}.
// => per lane: two contiguous 8-half (16B) chunks at k0+(lane/16)*8 and +16.
// ---------------------------------------------------------------------------
__device__ __forceinline__ v16h load_afrag(const _Float16* __restrict__ base,
                                           int row_stride, int m0, int k0, int lane) {
  const int r  = m0 + (lane & 15);
  const int kk = k0 + ((lane >> 4) << 3);
  const _Float16* p0 = base + (size_t)r * row_stride + kk;
  v8h lo = *(const v8h*)(p0);
  v8h hi = *(const v8h*)(p0 + 16);
  v16h a;
#pragma unroll
  for (int i = 0; i < 8; ++i) { a[i] = lo[i]; a[i + 8] = hi[i]; }
  return a;
}

// ---------------------------------------------------------------------------
// Device-wide sense-reversing barrier (32 resident blocks).
// bar[0] = arrival counter, bar[1] = generation.
// ---------------------------------------------------------------------------
__device__ __forceinline__ void grid_sync(unsigned* bar, unsigned nblocks) {
  __threadfence();          // make this thread's global stores visible (agent)
  __syncthreads();
  if (threadIdx.x == 0) {
    unsigned gen = __hip_atomic_load(&bar[1], __ATOMIC_RELAXED, __HIP_MEMORY_SCOPE_AGENT);
    unsigned a = __hip_atomic_fetch_add(&bar[0], 1u, __ATOMIC_ACQ_REL, __HIP_MEMORY_SCOPE_AGENT);
    if (a == nblocks - 1u) {
      __hip_atomic_store(&bar[0], 0u, __ATOMIC_RELAXED, __HIP_MEMORY_SCOPE_AGENT);
      __hip_atomic_fetch_add(&bar[1], 1u, __ATOMIC_RELEASE, __HIP_MEMORY_SCOPE_AGENT);
    } else {
      while (__hip_atomic_load(&bar[1], __ATOMIC_ACQUIRE, __HIP_MEMORY_SCOPE_AGENT) == gen) {
        __builtin_amdgcn_s_sleep(2);
      }
    }
  }
  __syncthreads();
  __threadfence();          // acquire side: don't read stale h through L0/WGP$
}

// ---------------------------------------------------------------------------
// fp32 -> f16 elementwise cast (x is already [b*T+t][IN] row-major when flat).
// ---------------------------------------------------------------------------
__global__ void k_cast_f16(const float* __restrict__ src, _Float16* __restrict__ dst, int n) {
  int i = blockIdx.x * blockDim.x + threadIdx.x;
  if (i < n) dst[i] = (_Float16)src[i];
}

// ---------------------------------------------------------------------------
// Pack W [H][K] (fp32, row n = output neuron) into WMMA B-fragment layout:
// flat = ((n16*(K/32) + ki)*32 + lane)*16 + e,
// with n = n16*16 + lane%16, k = ki*32 + (lane/16)*8 + (e/8)*16 + e%8  (B[k][n] = W[n][k]).
// ---------------------------------------------------------------------------
__global__ void k_pack_w(const float* __restrict__ W, _Float16* __restrict__ P, int K, int total) {
  int i = blockIdx.x * blockDim.x + threadIdx.x;
  if (i >= total) return;
  int e    = i & 15;
  int l    = (i >> 4) & 31;
  int frag = i >> 9;
  int KI   = K >> 5;
  int ki   = frag % KI;
  int n16  = frag / KI;
  int n = n16 * 16 + (l & 15);
  int k = ki * 32 + ((l >> 4) << 3) + ((e >> 3) << 4) + (e & 7);
  P[i] = (_Float16)W[(size_t)n * K + k];
}

// ---------------------------------------------------------------------------
// Input projection GEMM: for m = b*T+t,
//   XW[t][b][n] = sum_k A[m][k]*W[n][k] + (b1[n]+b2[n])   (t-major output!)
// Each wave computes a 16x64 tile (4 accumulators). Grid: (M/16)*(H/64)/8 blocks.
// ---------------------------------------------------------------------------
__global__ void __launch_bounds__(256) k_proj(const _Float16* __restrict__ A,
                                              const _Float16* __restrict__ BP,
                                              const float* __restrict__ b1,
                                              const float* __restrict__ b2,
                                              _Float16* __restrict__ XW, int K) {
  const int lane  = threadIdx.x & 31;
  const int wid   = threadIdx.x >> 5;
  const int wtile = blockIdx.x * 8 + wid;
  const int nG    = wtile & 15;        // H/64 = 16 column groups
  const int mTile = wtile >> 4;        // M/16 row tiles
  const int m0 = mTile * 16;
  const int n0 = nG * 64;
  const int KI = K >> 5;

  v8f z = {};
  v8f acc[4];
#pragma unroll
  for (int i = 0; i < 4; ++i) acc[i] = z;

  for (int ki = 0; ki < KI; ++ki) {
    v16h a = load_afrag(A, K, m0, ki * 32, lane);
#pragma unroll
    for (int nt = 0; nt < 4; ++nt) {
      size_t frag = (size_t)((n0 >> 4) + nt) * KI + ki;
      v16h bf = *(const v16h*)(BP + frag * 512 + lane * 16);
      acc[nt] = __builtin_amdgcn_wmma_f32_16x16x32_f16(false, a, false, bf,
                                                       (short)0, acc[nt], false, false);
    }
  }
#pragma unroll
  for (int nt = 0; nt < 4; ++nt) {
    int n = n0 + nt * 16 + (lane & 15);
    float bias = b1[n] + b2[n];
#pragma unroll
    for (int r = 0; r < 8; ++r) {
      int m = m0 + r + ((lane >> 4) << 3);
      int b = m / T_;
      int t = m % T_;                    // tile rows share b; t = t0..t0+15
      XW[((size_t)t * B_ + b) * H_ + n] = (_Float16)(acc[nt][r] + bias);
    }
  }
}

// ---------------------------------------------------------------------------
// Init: zero both h buffers and the barrier state (run before each recurrence).
// ---------------------------------------------------------------------------
__global__ void k_init(_Float16* h0, _Float16* h1, unsigned* bar) {
  int i = blockIdx.x * blockDim.x + threadIdx.x;
  if (i < B_ * H_) { h0[i] = (_Float16)0.f; h1[i] = (_Float16)0.f; }
  if (i < 2) bar[i] = 0u;
}

// ---------------------------------------------------------------------------
// Persistent recurrence kernel. 32 blocks x 256 threads (8 waves).
// Block nb owns output columns [nb*32, nb*32+32); its W_hh B-fragments
// (64KB contiguous in the packed buffer) are staged once into LDS.
// Wave w: m-tile = w%4 (rows of B=64), n-tile = w/4 (16 cols each).
// Per step t: acc = xw[t] tile; 32 WMMA k-iterations over h_{t-1};
// h_t = tanh(acc); write to the other h buffer; grid barrier.
// ---------------------------------------------------------------------------
__global__ void __launch_bounds__(256, 1) k_rnn_rec(const _Float16* __restrict__ XW,
                                                    const _Float16* __restrict__ WP,
                                                    _Float16* __restrict__ h0buf,
                                                    _Float16* __restrict__ h1buf,
                                                    _Float16* __restrict__ YS,
                                                    float* __restrict__ HLAST,
                                                    unsigned* __restrict__ bar,
                                                    int store_ys, int layer) {
  __shared__ _Float16 ldsW[32768];   // 64 KB: 64 fragments x 512 halves
  const int nb   = blockIdx.x;
  const int tid  = threadIdx.x;
  const int lane = tid & 31;
  const int wid  = tid >> 5;

  { // stage this block's 64KB weight slice into LDS
    const v4i* src = (const v4i*)(WP + (size_t)nb * 64 * 512);
    v4i* dst = (v4i*)ldsW;
#if __has_builtin(__builtin_amdgcn_global_load_async_to_lds_b128) && \
    __has_builtin(__builtin_amdgcn_s_wait_asynccnt)
    // CDNA5 async-to-LDS datapath (tracked by ASYNCcnt)
    for (int i = tid; i < 4096; i += 256) {
      __builtin_amdgcn_global_load_async_to_lds_b128(
          (GLOBAL_AS v4i*)(src + i), (LDS_AS v4i*)(dst + i), 0, 0);
    }
    __builtin_amdgcn_s_wait_asynccnt(0);
#else
    for (int i = tid; i < 4096; i += 256) dst[i] = src[i];
#endif
  }
  __syncthreads();

  const int mt = wid & 3;
  const int nt = wid >> 2;
  const int m0 = mt * 16;
  const int n  = nb * 32 + nt * 16 + (lane & 15);
  const int rbase = (lane >> 4) << 3;
  _Float16* hb[2] = { h0buf, h1buf };

  for (int t = 0; t < T_; ++t) {
    const _Float16* hprev = hb[t & 1];
    _Float16* hnext = hb[(t + 1) & 1];

    v8f acc;
#pragma unroll
    for (int r = 0; r < 8; ++r) {
      int b = m0 + r + rbase;
      acc[r] = (float)XW[((size_t)t * B_ + b) * H_ + n];
    }
    // prefetch next step's xw slab (overlaps the upcoming barrier wait)
    if (t + 1 < T_) {
      __builtin_prefetch(&XW[((size_t)(t + 1) * B_ + m0 + rbase) * H_ + n], 0, 1);
    }
#pragma unroll 4
    for (int ki = 0; ki < 32; ++ki) {
      v16h a  = load_afrag(hprev, H_, m0, ki * 32, lane);
      v16h bf = *(const v16h*)(ldsW + (nt * 32 + ki) * 512 + lane * 16);
      acc = __builtin_amdgcn_wmma_f32_16x16x32_f16(false, a, false, bf,
                                                   (short)0, acc, false, false);
    }
#pragma unroll
    for (int r = 0; r < 8; ++r) {
      int b = m0 + r + rbase;
      float hv = tanhf(acc[r]);
      hnext[(size_t)b * H_ + n] = (_Float16)hv;
      if (store_ys) YS[((size_t)b * T_ + t) * H_ + n] = (_Float16)hv;
      if (t == T_ - 1) HLAST[(size_t)b * 2 * H_ + layer * H_ + n] = hv;
    }
    grid_sync(bar, gridDim.x);
  }
}

// ---------------------------------------------------------------------------
// Final: out[b][o] = sigmoid(b_lin[o] + hlast[b][:] . W_lin[o][:]).  Tiny.
// ---------------------------------------------------------------------------
__global__ void k_final(const float* __restrict__ HL, const float* __restrict__ WL,
                        const float* __restrict__ bl, float* __restrict__ out) {
  int i = blockIdx.x * blockDim.x + threadIdx.x;   // 8192 = 64*128
  int o = i & (OUT_ - 1);
  int b = i >> 7;
  float acc = bl[o];
  const float* h = HL + (size_t)b * 2 * H_;
  const float* w = WL + (size_t)o * 2 * H_;
  for (int f = 0; f < 2 * H_; ++f) acc += h[f] * w[f];
  out[i] = 1.f / (1.f + expf(-acc));
}

// ---------------------------------------------------------------------------
extern "C" void kernel_launch(void* const* d_in, const int* in_sizes, int n_in,
                              void* d_out, int out_size, void* d_ws, size_t ws_size,
                              hipStream_t stream) {
  (void)in_sizes; (void)n_in; (void)out_size; (void)ws_size;
  const float* x    = (const float*)d_in[0];
  const float* Wih0 = (const float*)d_in[1];
  const float* Whh0 = (const float*)d_in[2];
  const float* bih0 = (const float*)d_in[3];
  const float* bhh0 = (const float*)d_in[4];
  const float* Wih1 = (const float*)d_in[5];
  const float* Whh1 = (const float*)d_in[6];
  const float* bih1 = (const float*)d_in[7];
  const float* bhh1 = (const float*)d_in[8];
  const float* Wlin = (const float*)d_in[9];
  const float* blin = (const float*)d_in[10];
  float* out = (float*)d_out;

  char* ws = (char*)d_ws;
  size_t off = 0;
  auto alloc = [&](size_t bytes) {
    char* p = ws + off;
    off = (off + bytes + 255) & ~(size_t)255;
    return p;
  };
  _Float16* xh    = (_Float16*)alloc((size_t)M_ * IN_ * 2);   // 16 MB
  _Float16* wih0p = (_Float16*)alloc((size_t)H_ * IN_ * 2);   // 512 KB
  _Float16* whh0p = (_Float16*)alloc((size_t)H_ * H_  * 2);   // 2 MB
  _Float16* wih1p = (_Float16*)alloc((size_t)H_ * H_  * 2);   // 2 MB
  _Float16* whh1p = (_Float16*)alloc((size_t)H_ * H_  * 2);   // 2 MB
  _Float16* xw    = (_Float16*)alloc((size_t)M_ * H_  * 2);   // 64 MB (reused by both layers)
  _Float16* ysh   = (_Float16*)alloc((size_t)M_ * H_  * 2);   // 64 MB
  _Float16* h0b   = (_Float16*)alloc((size_t)B_ * H_  * 2);
  _Float16* h1b   = (_Float16*)alloc((size_t)B_ * H_  * 2);
  float*    hlast = (float*)  alloc((size_t)B_ * 2 * H_ * 4);
  unsigned* bar   = (unsigned*)alloc(256);

  // 1) cast x to f16 (flat [b*T+t][IN] already)
  k_cast_f16<<<(M_ * IN_ + 255) / 256, 256, 0, stream>>>(x, xh, M_ * IN_);

  // 2) pack all weights into WMMA B-fragment layout
  k_pack_w<<<(H_ * IN_) / 256, 256, 0, stream>>>(Wih0, wih0p, IN_, H_ * IN_);
  k_pack_w<<<(H_ * H_) / 256, 256, 0, stream>>>(Whh0, whh0p, H_, H_ * H_);
  k_pack_w<<<(H_ * H_) / 256, 256, 0, stream>>>(Wih1, wih1p, H_, H_ * H_);
  k_pack_w<<<(H_ * H_) / 256, 256, 0, stream>>>(Whh1, whh1p, H_, H_ * H_);

  const int projGrid = (M_ / 16) * (H_ / 64) / 8;   // 4096 blocks

  // 3) layer 0: input projection, then persistent recurrence
  k_proj<<<projGrid, 256, 0, stream>>>(xh, wih0p, bih0, bhh0, xw, IN_);
  k_init<<<(B_ * H_ + 255) / 256, 256, 0, stream>>>(h0b, h1b, bar);
  k_rnn_rec<<<32, 256, 0, stream>>>(xw, whh0p, h0b, h1b, ysh, hlast, bar, 1, 0);

  // 4) layer 1: projection of ys0, then recurrence (no ys store)
  k_proj<<<projGrid, 256, 0, stream>>>(ysh, wih1p, bih1, bhh1, xw, H_);
  k_init<<<(B_ * H_ + 255) / 256, 256, 0, stream>>>(h0b, h1b, bar);
  k_rnn_rec<<<32, 256, 0, stream>>>(xw, whh1p, h0b, h1b, ysh, hlast, bar, 0, 1);

  // 5) final linear + sigmoid
  k_final<<<(B_ * OUT_) / 256, 256, 0, stream>>>(hlast, Wlin, blin, out);
}
